// MLPNetworkWithAttention2__80169859547547
// MI455X (gfx1250) — matrix operand order repaired and checked by reading, not compiled, not run on hardware
//
#include <hip/hip_runtime.h>

typedef __attribute__((ext_vector_type(16))) _Float16 v16h;
typedef __attribute__((ext_vector_type(8)))  _Float16 v8h;
typedef __attribute__((ext_vector_type(8)))  float    v8f;

#define NT 512          // threads per block = 16 waves
#define MROWS 32        // batch rows per block

// ---------------------------------------------------------------------------
// Prep: convert W (K x N, f32, row-major) -> Wt (N x Kpad, f16, row-major),
// transposed + zero-padded along K so each lane's WMMA B fragment is two contiguous 16B runs.
// ---------------------------------------------------------------------------
__global__ void convert_transpose_f16(const float* __restrict__ W,
                                      _Float16* __restrict__ Wt,
                                      int K, int N, int Kpad) {
  int i = blockIdx.x * blockDim.x + threadIdx.x;
  int total = N * Kpad;
  if (i >= total) return;
  int n = i / Kpad, k = i % Kpad;
  Wt[i] = (k < K) ? (_Float16)W[(size_t)k * N + n] : (_Float16)0.f;
}

// ---------------------------------------------------------------------------
// WMMA fragment layouts per CDNA5 ISA 7.12.2 (v_wmma_f32_16x16x32_f16):
//  A 16x32 f16 : lane<16 holds M=lane, K={0..7,16..23}; lane>=16: K +8
//  B 32x16 f16 : lane<16 holds N=lane, K={0..7,16..23}; lane>=16: K +8
//  C/D f32     : reg r of lane     -> M = r + 8*half,  N = lane&15
// ---------------------------------------------------------------------------
__device__ __forceinline__ v16h cat16(v8h lo, v8h hi) {
  v16h r;
#pragma unroll
  for (int i = 0; i < 8; ++i) { r[i] = lo[i]; r[i + 8] = hi[i]; }
  return r;
}

// Two 16x16 output tiles (rows 0..15 and 16..31, cols [n0,n0+16)) of
// A(32 x 32*kSteps) * Wt^T.  Each B fragment is loaded once and reused by
// both row tiles (halves L2 weight traffic vs one row tile).
__device__ __forceinline__ void gemm32(const _Float16* __restrict__ ldsA, int strideA,
                                       const _Float16* __restrict__ Wt, int Kpad,
                                       int n0, int kSteps, v8f& acc0, v8f& acc1) {
  const int lane = threadIdx.x & 31;
  const int half = lane >> 4;
  const int idx  = lane & 15;
  const _Float16* arow0 = ldsA + idx * strideA + 8 * half;
  const _Float16* arow1 = arow0 + 16 * strideA;
  const _Float16* bcol  = Wt + (size_t)(n0 + idx) * Kpad + 8 * half;
#pragma unroll
  for (int ks = 0; ks < kSteps; ++ks) {
    const int k0 = ks * 32;
    v8h b0 = *(const v8h*)(bcol + k0);
    v8h b1 = *(const v8h*)(bcol + k0 + 16);
    v8h a00 = *(const v8h*)(arow0 + k0);
    v8h a01 = *(const v8h*)(arow0 + k0 + 16);
    v8h a10 = *(const v8h*)(arow1 + k0);
    v8h a11 = *(const v8h*)(arow1 + k0 + 16);
    v16h B  = cat16(b0, b1);
    v16h A0 = cat16(a00, a01);
    v16h A1 = cat16(a10, a11);
    acc0 = __builtin_amdgcn_wmma_f32_16x16x32_f16(false, A0, false, B,
                                                  (short)0, acc0, false, false);
    acc1 = __builtin_amdgcn_wmma_f32_16x16x32_f16(false, A1, false, B,
                                                  (short)0, acc1, false, false);
  }
}

__device__ __forceinline__ void store_tiles(v8f acc0, v8f acc1,
                                            const float* __restrict__ bias, int n0,
                                            _Float16* __restrict__ ldsOut,
                                            int strideOut, bool relu) {
  const int lane = threadIdx.x & 31;
  const int half = lane >> 4;
  const int n = n0 + (lane & 15);
  const float bv = bias[n];
#pragma unroll
  for (int r = 0; r < 8; ++r) {
    float v0 = acc0[r] + bv;
    float v1 = acc1[r] + bv;
    if (relu) { v0 = fmaxf(v0, 0.f); v1 = fmaxf(v1, 0.f); }
    ldsOut[(size_t)(r + 8 * half) * strideOut + n]      = (_Float16)v0;
    ldsOut[(size_t)(r + 8 * half + 16) * strideOut + n] = (_Float16)v1;
  }
}

// f16 dot helper over 8 contiguous elements
__device__ __forceinline__ float dot8(const _Float16* a, const _Float16* b) {
  v8h va = *(const v8h*)a;
  v8h vb = *(const v8h*)b;
  float s = 0.f;
#pragma unroll
  for (int i = 0; i < 8; ++i) s += (float)va[i] * (float)vb[i];
  return s;
}

// ---------------------------------------------------------------------------
// Fused MLP + multi-head attention. One block = 32 batch rows, 16 waves.
// All activations live in LDS (~234 KB); weights are pre-transposed f16.
// ---------------------------------------------------------------------------
__global__ __launch_bounds__(NT) void fused_mlp_attn(
    const float* __restrict__ x,
    const float* __restrict__ b_fc1, const float* __restrict__ b_fc2,
    const float* __restrict__ b_enc, const float* __restrict__ b_heads,
    const float* __restrict__ b_dec, const float* __restrict__ b_fc3,
    const float* __restrict__ W_q,   const float* __restrict__ b_q,
    const int*   __restrict__ agent_idx,
    const _Float16* __restrict__ Wt_fc1, const _Float16* __restrict__ Wt_fc2,
    const _Float16* __restrict__ Wt_enc, const _Float16* __restrict__ Wt_heads,
    const _Float16* __restrict__ Wt_dec, const _Float16* __restrict__ Wt_fc3,
    float* __restrict__ out)
{
  extern __shared__ char smem_raw[];
  _Float16* sAx    = (_Float16*)smem_raw;      // 32 x 96  (enc input, pad 77->96)
  _Float16* sAdec  = sAx    + MROWS * 96;      // 32 x 32  (dec input, pad 10->32)
  _Float16* sMidE  = sAdec  + MROWS * 32;      // 32 x 256 encoder_input
  _Float16* sMidD  = sMidE  + MROWS * 256;     // 32 x 256 decoder_input
  _Float16* sEncH  = sMidD  + MROWS * 256;     // 32 x 256 relu(enc)
  _Float16* sDecH  = sEncH  + MROWS * 256;     // 32 x 256 relu(dec)
  _Float16* sCtx   = sDecH  + MROWS * 256;     // 32 x 256 context
  _Float16* sT1    = sCtx   + MROWS * 256;     // 32 x 256 relu(fc3)
  _Float16* sHeads = sT1    + MROWS * 256;     // 8 x 32 x 256 head activations
  float* sScore = (float*)(sHeads + 8 * MROWS * 256);  // 32 x 8
  float* sAttn  = sScore + MROWS * 8;                  // 32 x 8

  const int t    = threadIdx.x;
  const int wave = t >> 5;
  const int row0 = blockIdx.x * MROWS;
  const int a    = agent_idx[0];
  const int d0   = 5 * a;          // D = [0,5,10,15]; act_self = acts[:, 5a:5a+5]
  const int n0   = wave * 16;      // this wave's column tile

  // Stage activations (fp32 -> f16) with agent-dependent column gather.
  for (int i = t; i < MROWS * 96; i += NT) {
    int r = i / 96, c = i % 96;
    float v = 0.f;
    if (c < 77) {
      int col = (c < 72) ? c : (72 + d0 + (c - 72));      // obs ++ act_self
      v = x[(size_t)(row0 + r) * 87 + col];
    }
    sAx[i] = (_Float16)v;
  }
  for (int i = t; i < MROWS * 32; i += NT) {
    int r = i / 32, c = i % 32;
    float v = 0.f;
    if (c < 10) {
      int col = (c < d0) ? (72 + c) : (77 + c);           // act_others
      v = x[(size_t)(row0 + r) * 87 + col];
    }
    sAdec[i] = (_Float16)v;
  }
  __syncthreads();

  // Phase 1: encoder_input = Ax @ W_fc1 + b ; decoder_input = Adec @ W_fc2 + b
  {
    v8f e0 = {}, e1 = {}, d0v = {}, d1v = {};
    gemm32(sAx, 96, Wt_fc1, 96, n0, 3, e0, e1);
    gemm32(sAdec, 32, Wt_fc2, 32, n0, 1, d0v, d1v);
    store_tiles(e0, e1, b_fc1, n0, sMidE, 256, false);
    store_tiles(d0v, d1v, b_fc2, n0, sMidD, 256, false);
  }
  __syncthreads();

  // Phase 2: enc_h = relu(enc_in @ W_enc + b); decH = relu(dec_in @ W_dec + b)
  {
    v8f e0 = {}, e1 = {}, d0v = {}, d1v = {};
    gemm32(sMidE, 256, Wt_enc, 256, n0, 8, e0, e1);
    gemm32(sMidD, 256, Wt_dec, 256, n0, 8, d0v, d1v);
    store_tiles(e0, e1, b_enc, n0, sEncH, 256, true);
    store_tiles(d0v, d1v, b_dec, n0, sDecH, 256, true);
  }
  __syncthreads();

  // Phase 3: 8 attention heads, relu, resident in LDS.
  for (int h = 0; h < 8; ++h) {
    if (h < 7) {  // cover next head's B-stream startup latency
      __builtin_prefetch(Wt_heads + (size_t)(h + 1) * 256 * 256 +
                             (size_t)n0 * 256, 0, 3);
    }
    v8f c0 = {}, c1 = {};
    gemm32(sEncH, 256, Wt_heads + (size_t)h * 256 * 256, 256, n0, 8, c0, c1);
    store_tiles(c0, c1, b_heads + h * 256, n0, sHeads + (size_t)h * MROWS * 256,
                256, true);
  }
  __syncthreads();

  // scores[m][h] = <heads[h][m], decH[m]>  — 2 threads per (m,h), 128 elems each
  {
    const int m = t >> 4;            // 0..31
    const int h = (t >> 1) & 7;      // 0..7
    const int sub = t & 1;           // half of the 256-length dot
    const _Float16* hp = sHeads + (size_t)h * MROWS * 256 + m * 256 + sub * 128;
    const _Float16* dp = sDecH + m * 256 + sub * 128;
    float s = 0.f;
#pragma unroll
    for (int c = 0; c < 16; ++c) s += dot8(hp + c * 8, dp + c * 8);
    s += __shfl_xor(s, 1, 32);
    if (sub == 0) sScore[m * 8 + h] = s;
  }
  __syncthreads();
  // softmax over h
  if (t < MROWS) {
    const int m = t;
    float mx = sScore[m * 8];
#pragma unroll
    for (int h = 1; h < 8; ++h) mx = fmaxf(mx, sScore[m * 8 + h]);
    float e[8], sum = 0.f;
#pragma unroll
    for (int h = 0; h < 8; ++h) { e[h] = __expf(sScore[m * 8 + h] - mx); sum += e[h]; }
    const float inv = 1.f / sum;
#pragma unroll
    for (int h = 0; h < 8; ++h) sAttn[m * 8 + h] = e[h] * inv;
  }
  __syncthreads();
  // context[m][n] = sum_h heads[h][m][n] * attn[m][h]
  for (int i = t; i < MROWS * 256; i += NT) {
    const int m = i >> 8, n = i & 255;
    float c = 0.f;
#pragma unroll
    for (int h = 0; h < 8; ++h)
      c += (float)sHeads[(size_t)h * MROWS * 256 + m * 256 + n] * sAttn[m * 8 + h];
    sCtx[i] = (_Float16)c;
  }
  __syncthreads();
  // t1 = relu(ctx @ W_fc3 + b)
  {
    v8f c0 = {}, c1 = {};
    gemm32(sCtx, 256, Wt_fc3, 256, n0, 8, c0, c1);
    store_tiles(c0, c1, b_fc3, n0, sT1, 256, true);
  }
  __syncthreads();
  // out[m] = <t1[m], W_q> + b_q  — 16 threads per row, shuffle-tree reduce
  {
    const int m = t >> 4;            // 0..31
    const int chunk = t & 15;        // 16 elems each
    const _Float16* tp = sT1 + m * 256 + chunk * 16;
    const float* wq = W_q + chunk * 16;
    float s = 0.f;
#pragma unroll
    for (int i = 0; i < 16; ++i) s += (float)tp[i] * wq[i];
#pragma unroll
    for (int mask = 1; mask <= 8; mask <<= 1) s += __shfl_xor(s, mask, 32);
    if (chunk == 0) out[row0 + m] = s + b_q[0];
  }
}

// ---------------------------------------------------------------------------
extern "C" void kernel_launch(void* const* d_in, const int* in_sizes, int n_in,
                              void* d_out, int out_size, void* d_ws, size_t ws_size,
                              hipStream_t stream) {
  (void)in_sizes; (void)n_in; (void)ws_size;
  const float* x      = (const float*)d_in[0];
  const float* W_fc1  = (const float*)d_in[1];
  const float* b_fc1  = (const float*)d_in[2];
  const float* W_fc2  = (const float*)d_in[3];
  const float* b_fc2  = (const float*)d_in[4];
  const float* W_enc  = (const float*)d_in[5];
  const float* b_enc  = (const float*)d_in[6];
  const float* W_hds  = (const float*)d_in[7];
  const float* b_hds  = (const float*)d_in[8];
  const float* W_dec  = (const float*)d_in[9];
  const float* b_dec  = (const float*)d_in[10];
  const float* W_fc3  = (const float*)d_in[11];
  const float* b_fc3  = (const float*)d_in[12];
  const float* W_q    = (const float*)d_in[13];
  const float* b_q    = (const float*)d_in[14];
  const int*   agent  = (const int*)d_in[15];
  float* out = (float*)d_out;

  // Workspace: pre-transposed f16 weights.
  _Float16* Wt_fc1   = (_Float16*)d_ws;               // 256*96
  _Float16* Wt_fc2   = Wt_fc1 + 256 * 96;             // 256*32
  _Float16* Wt_enc   = Wt_fc2 + 256 * 32;             // 256*256
  _Float16* Wt_dec   = Wt_enc + 256 * 256;
  _Float16* Wt_fc3   = Wt_dec + 256 * 256;
  _Float16* Wt_heads = Wt_fc3 + 256 * 256;            // 8*256*256

  auto ct = [&](const float* W, _Float16* Wt, int K, int N, int Kpad) {
    const int total = N * Kpad;
    convert_transpose_f16<<<(total + 255) / 256, 256, 0, stream>>>(W, Wt, K, N, Kpad);
  };
  ct(W_fc1, Wt_fc1, 77, 256, 96);
  ct(W_fc2, Wt_fc2, 10, 256, 32);
  ct(W_enc, Wt_enc, 256, 256, 256);
  ct(W_dec, Wt_dec, 256, 256, 256);
  ct(W_fc3, Wt_fc3, 256, 256, 256);
  for (int h = 0; h < 8; ++h)
    ct(W_hds + (size_t)h * 256 * 256, Wt_heads + (size_t)h * 256 * 256, 256, 256, 256);

  const size_t shmem =
      (size_t)(MROWS * 96 + MROWS * 32 + 6 * MROWS * 256 + 8 * MROWS * 256) *
          sizeof(_Float16) +
      (size_t)2 * MROWS * 8 * sizeof(float);
  static bool attr_set = [&]() {
    hipFuncSetAttribute((const void*)fused_mlp_attn,
                        hipFuncAttributeMaxDynamicSharedMemorySize, (int)shmem);
    return true;
  }();
  (void)attr_set;

  const int rows = out_size;           // OUT == 1 column
  fused_mlp_attn<<<rows / MROWS, NT, shmem, stream>>>(
      x, b_fc1, b_fc2, b_enc, b_hds, b_dec, b_fc3, W_q, b_q, agent,
      Wt_fc1, Wt_fc2, Wt_enc, Wt_heads, Wt_dec, Wt_fc3, out);
}
